// Qwen3Attention_18038862643519
// MI455X (gfx1250) — matrix operand order repaired and checked by reading, not compiled, not run on hardware
//
#include <hip/hip_runtime.h>
#include <hip/hip_bf16.h>
#include <math.h>
#include <stdint.h>

// ---------------------------------------------------------------------------
// Qwen3 attention for MI455X (gfx1250, wave32).
// - bf16 WMMA (v_wmma_f32_16x16x32_bf16) for all GEMMs, f32 accumulate
// - fp32 softmax / RMSNorm / RoPE
// - flash attention (no materialized SxS score tensor)
// - GLOBAL_LOAD_ASYNC_TO_LDS_B128 (ASYNCcnt) for raw bf16 staging
// - V stored transposed [kv][dim][token] so ALL WMMA B-fragments are
//   contiguous-per-lane (2x ds_load_b128 instead of 16x ds_load_u16)
// ---------------------------------------------------------------------------

typedef __attribute__((ext_vector_type(16))) __bf16 v16bf;
typedef __attribute__((ext_vector_type(8)))  float  v8f;

#define S_LEN 2048
#define HID   4096
#define NH    32
#define NKV   8
#define DHEAD 128

static __device__ __forceinline__ v8f wmma_bf16(v16bf a, v16bf b, v8f c) {
  return __builtin_amdgcn_wmma_f32_16x16x32_bf16(false, a, false, b,
                                                 (short)0, c, false, false);
}

// CDNA5 async copy: 16B global -> LDS, tracked with ASYNCcnt (ISA 15.18.3).
// LDS offset = low 32 bits of the generic pointer (aperture truncation rule).
static __device__ __forceinline__ void async_copy_b128(void* lds, const void* g) {
  asm volatile("global_load_async_to_lds_b128 %0, %1, off"
               :: "v"((uint32_t)(uintptr_t)lds), "v"(g)
               : "memory");
}
static __device__ __forceinline__ void wait_async0() {
  asm volatile("s_wait_asynccnt 0x0" ::: "memory");
}

// A fragment: 16x32 tile, row-major in LDS, leading dim lda (ISA 7.12.2).
static __device__ __forceinline__ v16bf frag_A(const __bf16* p, int lda) {
  int lane = threadIdx.x & 31;
  int hl = lane >> 4, m = lane & 15;
  v16bf a;
#pragma unroll
  for (int e = 0; e < 16; ++e) {
    int k = (e & 7) + 8 * hl + 16 * (e >> 3);
    a[e] = p[m * lda + k];
  }
  return a;
}
// B fragment 32x16 where memory holds the tile [n][k]-major (Bt[n][k]):
// lane (kh,n) element e = p[n*ldb + (e&7) + 8*kh + 16*(e>>3)] -> two contiguous 16B runs.
static __device__ __forceinline__ v16bf frag_BT(const __bf16* p, int ldb) {
  int lane = threadIdx.x & 31;
  int kh = lane >> 4, n = lane & 15;
  v16bf b;
#pragma unroll
  for (int e = 0; e < 16; ++e) b[e] = p[n * ldb + (e & 7) + 8 * kh + 16 * (e >> 3)];
  return b;
}

// ---------------------------------------------------------------------------
// Kernel 1: fused QKV projection (+RMSNorm +RoPE). grid=(S/64, 48), block=256.
// KT = 64, 8 WMMA per wave per staging round.
// ---------------------------------------------------------------------------
__global__ void __launch_bounds__(256) qkv_kernel(
    const float* __restrict__ x, const float* __restrict__ Wq,
    const float* __restrict__ Wk, const float* __restrict__ Wv,
    const float* __restrict__ qnw, const float* __restrict__ knw,
    const int* __restrict__ pos_ids,
    __bf16* __restrict__ qbuf, __bf16* __restrict__ kbuf,
    __bf16* __restrict__ vtbuf)
{
  __shared__ __attribute__((aligned(16))) __bf16 sX[64 * 64];
  __shared__ __attribute__((aligned(16))) __bf16 sW[128 * 64];
  __shared__ __attribute__((aligned(16))) float  sY[64 * 128];

  const int tok0 = blockIdx.x * 64;
  const int hh   = blockIdx.y;     // 0..47
  const int tid  = threadIdx.x;
  const int w    = tid >> 5;       // 8 waves
  const int rs   = (w & 3) * 16;   // row strip
  const int cbo  = (w >> 2) * 64;  // column half

  const float* W; __bf16* dst; const float* nw; int mode;
  if (hh < NH) {
    W = Wq + (size_t)hh * DHEAD * HID;
    dst = qbuf + (size_t)hh * S_LEN * DHEAD; nw = qnw; mode = 0;
  } else if (hh < NH + NKV) {
    W = Wk + (size_t)(hh - NH) * DHEAD * HID;
    dst = kbuf + (size_t)(hh - NH) * S_LEN * DHEAD; nw = knw; mode = 1;
  } else {
    W = Wv + (size_t)(hh - NH - NKV) * DHEAD * HID;
    dst = vtbuf + (size_t)(hh - NH - NKV) * DHEAD * S_LEN;  // transposed
    nw = qnw; mode = 2;
  }

  const v8f zero8 = {0.f, 0.f, 0.f, 0.f, 0.f, 0.f, 0.f, 0.f};
  v8f acc[4];
#pragma unroll
  for (int t = 0; t < 4; ++t) acc[t] = zero8;

  const int xrow = tid >> 2, xkc = (tid & 3) * 16;   // X stage: 16 f32/thread
  const int wrow = tid >> 1, wkc = (tid & 1) * 32;   // W stage: 32 f32/thread

  for (int k0 = 0; k0 < HID; k0 += 64) {
    {   // stage X 64x64 (f32 -> bf16)
      const float* src = x + (size_t)(tok0 + xrow) * HID + k0 + xkc;
      if (k0 + 64 < HID) __builtin_prefetch(src + 64, 0, 1);
#pragma unroll
      for (int i = 0; i < 16; ++i) sX[xrow * 64 + xkc + i] = (__bf16)src[i];
    }
    {   // stage W 128x64 (f32 -> bf16)
      const float* src = W + (size_t)wrow * HID + k0 + wkc;
      if (k0 + 64 < HID) __builtin_prefetch(src + 64, 0, 1);
#pragma unroll
      for (int i = 0; i < 32; ++i) sW[wrow * 64 + wkc + i] = (__bf16)src[i];
    }
    __syncthreads();
    v16bf a0 = frag_A(sX + rs * 64, 64);
    v16bf a1 = frag_A(sX + rs * 64 + 32, 64);
    v16bf b[8];
#pragma unroll
    for (int t = 0; t < 4; ++t) b[t]     = frag_BT(sW + (cbo + t * 16) * 64, 64);
#pragma unroll
    for (int t = 0; t < 4; ++t) b[4 + t] = frag_BT(sW + (cbo + t * 16) * 64 + 32, 64);
#pragma unroll
    for (int t = 0; t < 4; ++t) acc[t] = wmma_bf16(a0, b[t], acc[t]);
#pragma unroll
    for (int t = 0; t < 4; ++t) acc[t] = wmma_bf16(a1, b[4 + t], acc[t]);
    __syncthreads();
  }

  {   // spill accumulators (C/D layout: lane(hl,n), vgpr r -> row r+8*hl)
    int lane = tid & 31, hl = lane >> 4, n = lane & 15;
#pragma unroll
    for (int t = 0; t < 4; ++t)
#pragma unroll
      for (int r = 0; r < 8; ++r)
        sY[(rs + r + 8 * hl) * 128 + cbo + t * 16 + n] = acc[t][r];
  }
  __syncthreads();

  if (mode == 2) {
    // V: transposed store. One thread per dim, 64 contiguous tokens each.
    if (tid < 128) {
#pragma unroll 8
      for (int i = 0; i < 64; ++i)
        dst[(size_t)tid * S_LEN + tok0 + i] = (__bf16)sY[i * 128 + tid];
    }
  } else if (tid < 64) {   // Q/K: one thread per token, RMSNorm + RoPE
    int tok = tok0 + tid;
    const float* y = sY + tid * 128;
    float ss = 0.f;
#pragma unroll 4
    for (int d = 0; d < 128; ++d) { float v = y[d]; ss += v * v; }
    float inv = rsqrtf(ss * (1.0f / 128.0f) + 1e-6f);
    float pos = (float)pos_ids[tok];
#pragma unroll 2
    for (int d = 0; d < 64; ++d) {
      float a = y[d]      * inv * nw[d];
      float b = y[d + 64] * inv * nw[d + 64];
      // inv_freq[d] = THETA^(-2d/128) = exp(-d * ln(1e6)/64)
      float invf = __expf(-(float)d * (13.815510557964274f / 64.0f));
      float ang = pos * invf;
      float c = __cosf(ang), s = __sinf(ang);
      dst[(size_t)tok * DHEAD + d]      = (__bf16)(a * c - b * s);
      dst[(size_t)tok * DHEAD + d + 64] = (__bf16)(b * c + a * s);
    }
  }
}

// ---------------------------------------------------------------------------
// Kernel 2: causal flash attention with GQA. grid=(S/64, 32), block=128.
// K/V/Q blocks staged with GLOBAL_LOAD_ASYNC_TO_LDS_B128. V is transposed
// so both QK^T and PV use the contiguous-per-lane B fragment path.
// ---------------------------------------------------------------------------
__global__ void __launch_bounds__(128) attn_kernel(
    const __bf16* __restrict__ qbuf, const __bf16* __restrict__ kbuf,
    const __bf16* __restrict__ vtbuf, __bf16* __restrict__ ctx)
{
  __shared__ __attribute__((aligned(16))) __bf16 sK[64 * 128];   // [key][dim]
  __shared__ __attribute__((aligned(16))) __bf16 sV[128 * 64];   // [dim][key]
  __shared__ __attribute__((aligned(16))) __bf16 sP[4][16 * 64];

  const int qb  = blockIdx.x;
  const int h   = blockIdx.y;
  const int kvh = h >> 2;          // H/KV = 4
  const int tid = threadIdx.x;
  const int w   = tid >> 5;
  const int lane = tid & 31, hl = lane >> 4, n16 = lane & 15;
  const int q0w = qb * 64 + w * 16;

  // stage the Q block (64x128 bf16 = 16KB) into sK via async copy
  {
    const __bf16* Qg = qbuf + ((size_t)h * S_LEN + (size_t)qb * 64) * DHEAD;
#pragma unroll
    for (int t = 0; t < 8; ++t) {
      int u = tid + t * 128;   // 16B unit index, 0..1023
      async_copy_b128(sK + u * 8, Qg + u * 8);
    }
    wait_async0();
    __syncthreads();
  }
  // Q fragments (A layout) from LDS
  v16bf aq[4];
#pragma unroll
  for (int kc = 0; kc < 4; ++kc)
#pragma unroll
    for (int e = 0; e < 16; ++e) {
      int k = kc * 32 + (e & 7) + 8 * hl + 16 * (e >> 3);
      aq[kc][e] = sK[(w * 16 + n16) * 128 + k];
    }

  const v8f zero8 = {0.f, 0.f, 0.f, 0.f, 0.f, 0.f, 0.f, 0.f};
  v8f O[8];
#pragma unroll
  for (int t = 0; t < 8; ++t) O[t] = zero8;
  float mrun[8], lrun[8];
#pragma unroll
  for (int r = 0; r < 8; ++r) { mrun[r] = -3.0e38f; lrun[r] = 0.f; }

  const float sm_scale = 0.08838834764831845f;  // 1/sqrt(128)

  for (int j = 0; j <= qb; ++j) {
    __syncthreads();   // protect sK (Q frags / previous K) and sV
    {  // async-stage K block [64 keys][128 dims] and Vt block [128 dims][64 keys]
      const __bf16* Kg  = kbuf + ((size_t)kvh * S_LEN + (size_t)j * 64) * DHEAD;
      const __bf16* Vtg = vtbuf + (size_t)kvh * DHEAD * S_LEN + (size_t)j * 64;
#pragma unroll
      for (int t = 0; t < 8; ++t) {
        int u = tid + t * 128;         // 0..1023 16B units
        async_copy_b128(sK + u * 8, Kg + u * 8);
        int d = u >> 3, c = (u & 7) * 8;   // Vt: 128 rows x 64 keys
        async_copy_b128(sV + d * 64 + c, Vtg + (size_t)d * S_LEN + c);
      }
      wait_async0();
      __syncthreads();
    }

    // S = Q K^T : batch 4 B-frags per K-chunk, 4 independent WMMA chains
    v8f s[4];
#pragma unroll
    for (int kt = 0; kt < 4; ++kt) s[kt] = zero8;
#pragma unroll
    for (int kc = 0; kc < 4; ++kc) {
      v16bf bk[4];
#pragma unroll
      for (int kt = 0; kt < 4; ++kt)
        bk[kt] = frag_BT(sK + (kt * 16) * 128 + kc * 32, 128);
#pragma unroll
      for (int kt = 0; kt < 4; ++kt)
        s[kt] = wmma_bf16(aq[kc], bk[kt], s[kt]);
    }

    // scale + causal mask + online softmax (rows live in 16-lane halves)
    float p[4][8];
#pragma unroll
    for (int r = 0; r < 8; ++r) {
      int query = q0w + r + 8 * hl;
      float loc = -3.0e38f;
#pragma unroll
      for (int kt = 0; kt < 4; ++kt) {
        int key = j * 64 + kt * 16 + n16;
        float sv = s[kt][r] * sm_scale;
        if (key > query) sv = -3.0e38f;
        p[kt][r] = sv;
        loc = fmaxf(loc, sv);
      }
#pragma unroll
      for (int m = 1; m <= 8; m <<= 1) loc = fmaxf(loc, __shfl_xor(loc, m, 32));
      float mnew  = fmaxf(mrun[r], loc);
      float scale = __expf(mrun[r] - mnew);
      float rsum = 0.f;
#pragma unroll
      for (int kt = 0; kt < 4; ++kt) {
        float e = __expf(p[kt][r] - mnew);
        p[kt][r] = e; rsum += e;
      }
#pragma unroll
      for (int m = 1; m <= 8; m <<= 1) rsum += __shfl_xor(rsum, m, 32);
      lrun[r] = lrun[r] * scale + rsum;
      mrun[r] = mnew;
#pragma unroll
      for (int t = 0; t < 8; ++t) O[t][r] *= scale;
    }

    // P: D-layout regs -> per-wave LDS -> A-layout regs (in-order DS per wave)
    __bf16* pw = sP[w];
#pragma unroll
    for (int kt = 0; kt < 4; ++kt)
#pragma unroll
      for (int r = 0; r < 8; ++r)
        pw[(r + 8 * hl) * 64 + kt * 16 + n16] = (__bf16)p[kt][r];

    v16bf ap[2];
#pragma unroll
    for (int kc = 0; kc < 2; ++kc)
#pragma unroll
      for (int e = 0; e < 16; ++e) {
        int k = kc * 32 + (e & 7) + 8 * hl + 16 * (e >> 3);
        ap[kc][e] = pw[n16 * 64 + k];
      }

    // O += P V : Vt tile is [dim][key] -> contiguous frag_BT loads
#pragma unroll
    for (int kc = 0; kc < 2; ++kc)
#pragma unroll
      for (int g = 0; g < 2; ++g) {
        v16bf bv[4];
#pragma unroll
        for (int d = 0; d < 4; ++d)
          bv[d] = frag_BT(sV + ((g * 4 + d) * 16) * 64 + kc * 32, 64);
#pragma unroll
        for (int d = 0; d < 4; ++d)
          O[g * 4 + d] = wmma_bf16(ap[kc], bv[d], O[g * 4 + d]);
      }
  }

  // normalize and write ctx[token][h*128 + d] (bf16)
#pragma unroll
  for (int r = 0; r < 8; ++r) {
    float invl = 1.0f / lrun[r];
    int tok = q0w + r + 8 * hl;
#pragma unroll
    for (int t = 0; t < 8; ++t)
      ctx[(size_t)tok * (NH * DHEAD) + h * DHEAD + t * 16 + n16] =
          (__bf16)(O[t][r] * invl);
  }
}

// ---------------------------------------------------------------------------
// Kernel 3: out = ctx @ Wo^T (fp32 out). grid=(S/64, HID/128), block=256.
// ctx tiles async-staged (already bf16); Wo converted through VGPRs.
// ---------------------------------------------------------------------------
__global__ void __launch_bounds__(256) out_kernel(
    const __bf16* __restrict__ ctx, const float* __restrict__ Wo,
    float* __restrict__ out)
{
  __shared__ __attribute__((aligned(16))) __bf16 sX[64 * 64];
  __shared__ __attribute__((aligned(16))) __bf16 sW[128 * 64];

  const int tok0 = blockIdx.x * 64;
  const int col0 = blockIdx.y * 128;
  const int tid  = threadIdx.x;
  const int w    = tid >> 5;
  const int rs   = (w & 3) * 16;
  const int cbo  = (w >> 2) * 64;

  const v8f zero8 = {0.f, 0.f, 0.f, 0.f, 0.f, 0.f, 0.f, 0.f};
  v8f acc[4];
#pragma unroll
  for (int t = 0; t < 4; ++t) acc[t] = zero8;

  const int xrow = tid >> 2, xkc = (tid & 3) * 16;
  const int wrow = tid >> 1, wkc = (tid & 1) * 32;

  for (int k0 = 0; k0 < HID; k0 += 64) {
    {   // async-stage ctx tile 64x64 bf16 (two 16B units per thread)
      const __bf16* src = ctx + (size_t)(tok0 + xrow) * HID + k0 + xkc;
      async_copy_b128(sX + xrow * 64 + xkc,     src);
      async_copy_b128(sX + xrow * 64 + xkc + 8, src + 8);
    }
    {   // stage Wo 128x64 (f32 -> bf16)
      const float* src = Wo + (size_t)(col0 + wrow) * HID + k0 + wkc;
      if (k0 + 64 < HID) __builtin_prefetch(src + 64, 0, 1);
#pragma unroll
      for (int i = 0; i < 32; ++i) sW[wrow * 64 + wkc + i] = (__bf16)src[i];
    }
    wait_async0();
    __syncthreads();
    v16bf a0 = frag_A(sX + rs * 64, 64);
    v16bf a1 = frag_A(sX + rs * 64 + 32, 64);
    v16bf b[8];
#pragma unroll
    for (int t = 0; t < 4; ++t) b[t]     = frag_BT(sW + (cbo + t * 16) * 64, 64);
#pragma unroll
    for (int t = 0; t < 4; ++t) b[4 + t] = frag_BT(sW + (cbo + t * 16) * 64 + 32, 64);
#pragma unroll
    for (int t = 0; t < 4; ++t) acc[t] = wmma_bf16(a0, b[t], acc[t]);
#pragma unroll
    for (int t = 0; t < 4; ++t) acc[t] = wmma_bf16(a1, b[4 + t], acc[t]);
    __syncthreads();
  }

  int lane = tid & 31, hl = lane >> 4, n = lane & 15;
#pragma unroll
  for (int t = 0; t < 4; ++t)
#pragma unroll
    for (int r = 0; r < 8; ++r)
      out[(size_t)(tok0 + rs + r + 8 * hl) * HID + col0 + cbo + t * 16 + n] =
          acc[t][r];
}

// ---------------------------------------------------------------------------
extern "C" void kernel_launch(void* const* d_in, const int* in_sizes, int n_in,
                              void* d_out, int out_size, void* d_ws, size_t ws_size,
                              hipStream_t stream) {
  (void)in_sizes; (void)n_in; (void)out_size; (void)ws_size;
  const float* x   = (const float*)d_in[0];
  // d_in[1]: attention_mask — causal mask recomputed arithmetically in-kernel
  const int*   pid = (const int*)d_in[2];
  const float* Wq  = (const float*)d_in[3];
  const float* Wk  = (const float*)d_in[4];
  const float* Wv  = (const float*)d_in[5];
  const float* Wo  = (const float*)d_in[6];
  const float* qnw = (const float*)d_in[7];
  const float* knw = (const float*)d_in[8];
  float* out = (float*)d_out;

  // workspace: Q 16MB | K 4MB | Vt 4MB | ctx 16MB  (bf16)
  __bf16* qbuf  = (__bf16*)d_ws;
  __bf16* kbuf  = qbuf + (size_t)NH  * S_LEN * DHEAD;
  __bf16* vtbuf = kbuf + (size_t)NKV * S_LEN * DHEAD;
  __bf16* ctx   = vtbuf + (size_t)NKV * S_LEN * DHEAD;

  qkv_kernel<<<dim3(S_LEN / 64, NH + 2 * NKV), 256, 0, stream>>>(
      x, Wq, Wk, Wv, qnw, knw, pid, qbuf, kbuf, vtbuf);
  attn_kernel<<<dim3(S_LEN / 64, NH), 128, 0, stream>>>(qbuf, kbuf, vtbuf, ctx);
  out_kernel<<<dim3(S_LEN / 64, HID / 128), 256, 0, stream>>>(ctx, Wo, out);
}
